// ConditionalCNF_27590869910221
// MI455X (gfx1250) — matrix operand order repaired and checked by reading, not compiled, not run on hardware
//
#include <hip/hip_runtime.h>
#include <hip/hip_bf16.h>

typedef __attribute__((ext_vector_type(16))) unsigned short v16u;
typedef __attribute__((ext_vector_type(16))) __bf16         v16bf;
typedef __attribute__((ext_vector_type(8)))  float          v8f;

#define HIDN 192
#define DD   16
#define STRA 200   // activation row stride (ushorts): 400B = 16B aligned, bank-spread

__device__ __forceinline__ unsigned short f2bf(float f){
  unsigned u = __builtin_bit_cast(unsigned, f);
  u += 0x7FFFu + ((u >> 16) & 1u);           // round-to-nearest-even
  return (unsigned short)(u >> 16);
}

__device__ __forceinline__ v8f wmma_bf16(v16bf a, v16bf b, v8f c){
  return __builtin_amdgcn_wmma_f32_16x16x32_bf16(false, a, false, b, (short)0, c, false, false);
}

// B-fragment / pre-permuted fragment: 16 contiguous ushorts per lane (2x b128)
__device__ __forceinline__ v16bf ldB16(const unsigned short* p){
  v16u u;
#pragma unroll
  for (int i = 0; i < 16; ++i) u[i] = p[i];
  return __builtin_bit_cast(v16bf, u);
}

// A-fragment from row-major [m][k] storage: lane holds row m=lane&15,
// K = ko + khalf + {0..7, 16..23}  (khalf = 8 for lanes 16..31)
__device__ __forceinline__ v16bf ldA16(const unsigned short* p){
  v16u u;
#pragma unroll
  for (int i = 0; i < 8;  ++i) u[i] = p[i];
#pragma unroll
  for (int i = 8; i < 16; ++i) u[i] = p[i + 8];
  return __builtin_bit_cast(v16bf, u);
}

__device__ __forceinline__ void gelu_both(float x, float& g, float& dg){
  // Phi(x) ~= sigmoid(k1*x + k3*x^3); matches erf-GELU to ~1e-3 (bf16-noise level)
  const float k1 = 1.5957691216057308f;
  const float k3 = 0.0713548162726009f;
  float x2 = x * x;
  float z  = x * fmaf(k3, x2, k1);
  float e  = __expf(-z);
  float s  = __builtin_amdgcn_rcpf(1.0f + e);
  g  = x * s;
  dg = s + g * (1.0f - s) * fmaf(3.0f * k3, x2, k1);
}

// ---------------- kernel 0: permute weights into per-lane WMMA fragment order -------------
__global__ __launch_bounds__(256) void cnf_permute(const float* __restrict__ W1,
                                                   const float* __restrict__ W2,
                                                   unsigned short* __restrict__ W1hf,
                                                   unsigned short* __restrict__ W2f){
  const int NT1 = 96 * 32 * 16;   // W1h: 12 n-tiles x 8 k-tiles
  const int NT2 = 72 * 32 * 16;   // W2 : 12 n-tiles x 6 k-tiles
  for (int idx = blockIdx.x * blockDim.x + threadIdx.x; idx < NT1 + NT2;
       idx += gridDim.x * blockDim.x){
    if (idx < NT1){
      int i = idx & 15, l = (idx >> 4) & 31, fragId = idx >> 9;
      int kt = fragId & 7, j = fragId >> 3;
      int k  = kt * 32 + (i & 7) + ((i >> 3) << 4) + ((l >= 16) ? 8 : 0);          // 0..255 -> W1 rows 16..271
      W1hf[idx] = f2bf(W1[(size_t)(16 + k) * HIDN + (j << 4) + (l & 15)]);
    } else {
      int q = idx - NT1;
      int i = q & 15, l = (q >> 4) & 31, fragId = q >> 9;
      int kt = fragId % 6, j = fragId / 6;
      int k  = kt * 32 + (i & 7) + ((i >> 3) << 4) + ((l >= 16) ? 8 : 0);
      W2f[q] = f2bf(W2[(size_t)k * HIDN + (j << 4) + (l & 15)]);
    }
  }
}

// ---------------- kernel 1: c = h @ W1[16:272] + b1  (time-invariant context) -------------
__global__ __launch_bounds__(256) void cnf_context(const float* __restrict__ h,
                                                   const float* __restrict__ b1,
                                                   const unsigned short* __restrict__ W1hf,
                                                   float* __restrict__ c){
  const int tid = threadIdx.x, w = tid >> 5, lane = tid & 31;
  const int rowbase = blockIdx.x * 128 + w * 16;
  const int n = lane & 15, mh = (lane >> 4) << 3;
  v8f acc[12];
#pragma unroll
  for (int j = 0; j < 12; ++j){
    float bv = b1[(j << 4) + n];
#pragma unroll
    for (int r = 0; r < 8; ++r) acc[j][r] = bv;
  }
#pragma unroll
  for (int kt = 0; kt < 8; ++kt){
    const float* hp = h + (size_t)(rowbase + n) * 256 + kt * 32 + mh;
    v16u au;
#pragma unroll
    for (int i = 0; i < 8;  ++i) au[i] = f2bf(hp[i]);
#pragma unroll
    for (int i = 8; i < 16; ++i) au[i] = f2bf(hp[i + 8]);
    v16bf a = __builtin_bit_cast(v16bf, au);
#pragma unroll
    for (int j = 0; j < 12; ++j){
      v16bf b = ldB16(W1hf + (size_t)(((j << 3) + kt) * 32 + lane) * 16);
      acc[j] = wmma_bf16(a, b, acc[j]);
    }
  }
#pragma unroll
  for (int j = 0; j < 12; ++j)
#pragma unroll
    for (int r = 0; r < 8; ++r)
      c[(size_t)(rowbase + r + mh) * HIDN + (j << 4) + n] = acc[j][r];
}

// ---------------- kernel 2: RK4 time integration with fused MLP+JVP -----------------------
__global__ __launch_bounds__(128) void cnf_main(const float* __restrict__ theta,
                                                const float* __restrict__ eps,
                                                const float* __restrict__ W1,
                                                const float* __restrict__ b2,
                                                const float* __restrict__ W3,
                                                const float* __restrict__ b3,
                                                const float* __restrict__ c,
                                                const unsigned short* __restrict__ W2f,
                                                float* __restrict__ out, int B){
  __shared__ __attribute__((aligned(16))) unsigned short W2fL[72 * 32 * 16];  // 73728 B
  __shared__ __attribute__((aligned(16))) unsigned short W1fL[12 * 32 * 16];  // 12288 B
  __shared__ __attribute__((aligned(16))) unsigned short W3fL[6 * 32 * 16];   //  6144 B
  __shared__ float w1tL[HIDN], b2L[HIDN], b3L[DD];
  __shared__ __attribute__((aligned(16))) unsigned short ybs[4][16 * 32];
  __shared__ __attribute__((aligned(16))) unsigned short ebs[4][16 * 32];
  // a1/ta1 double as a2/ta2: after the layer-2 A-fragments are hoisted to VGPRs,
  // the layer-1 activations in LDS are dead and layer-2 outputs reuse the space.
  __shared__ __attribute__((aligned(16))) unsigned short a1s[4][16 * STRA];   // 25600 B
  __shared__ __attribute__((aligned(16))) unsigned short ta1s[4][16 * STRA];  // 25600 B
  __shared__ float reds[4][16];
  // total LDS: 153,408 B

  const int tid = threadIdx.x;
  // W2 fragments: straight contiguous copy of the pre-permuted array (b128 pairs)
  {
    const uint4* src = (const uint4*)W2f;
    uint4*       dst = (uint4*)W2fL;
    for (int idx = tid; idx < (72 * 32 * 16) / 8; idx += 128) dst[idx] = src[idx];
  }
  // W1y rows 0..15 as B-fragments, K 16..31 zero-padded
  for (int idx = tid; idx < 12 * 32 * 16; idx += 128){
    int i = idx & 15, l = (idx >> 4) & 31, j = idx >> 9;
    int k = (i & 7) + ((i >> 3) << 4) + ((l >= 16) ? 8 : 0);
    W1fL[idx] = (k < 16) ? f2bf(W1[(size_t)k * HIDN + (j << 4) + (l & 15)]) : (unsigned short)0;
  }
  for (int idx = tid; idx < 6 * 32 * 16; idx += 128){
    int i = idx & 15, l = (idx >> 4) & 31, kt = idx >> 9;
    int k = kt * 32 + (i & 7) + ((i >> 3) << 4) + ((l >= 16) ? 8 : 0);
    W3fL[idx] = f2bf(W3[(size_t)k * DD + (l & 15)]);
  }
  for (int idx = tid; idx < HIDN; idx += 128){
    w1tL[idx] = W1[(size_t)272 * HIDN + idx];   // t-column of W1
    b2L[idx]  = b2[idx];
  }
  if (tid < DD) b3L[tid] = b3[tid];
  for (int idx = tid; idx < 4 * 16 * 32; idx += 128){   // zero K=16..31 pad of y/e tiles
    ((unsigned short*)ybs)[idx] = 0;
    ((unsigned short*)ebs)[idx] = 0;
  }
  __syncthreads();

  const int w = tid >> 5, lane = tid & 31;
  const int n  = lane & 15;
  const int mh = (lane >> 4) << 3;             // row offset for C/D layout, K-half for A
  const int rowbase = blockIdx.x * 64 + w * 16;

  unsigned short* yb  = ybs[w];
  unsigned short* eb  = ebs[w];
  unsigned short* ab  = a1s[w];
  unsigned short* tab = ta1s[w];
  float*          red = reds[w];
  const float*    cg  = c + (size_t)rowbase * HIDN;

  float y[8], er[8], ys[8], fo[8], f1[8], f2s[8], f3[8], fsum[8];
#pragma unroll
  for (int r = 0; r < 8; ++r){
    y[r] = theta[(size_t)(rowbase + r + mh) * DD + n];
    f1[r] = f2s[r] = f3[r] = 0.0f;
  }
  float lp = 0.0f;
  const float dtv = -0.1f;

  auto vf = [&](float t, float& ln){
#pragma unroll
    for (int r = 0; r < 8; ++r){
      yb[(r + mh) * 32 + n] = f2bf(ys[r]);
      eb[(r + mh) * 32 + n] = f2bf(er[r]);
    }
    v16bf aY = ldA16(yb + n * 32 + mh);
    v16bf aE = ldA16(eb + n * 32 + mh);
    // ---- layer 1: z1 = y@W1y + t*w1t + c ; tz1 = e@W1y
#pragma unroll
    for (int j = 0; j < 12; ++j){
      const int col = (j << 4) + n;
      v8f accP, accT;
      const float tw = t * w1tL[col];
#pragma unroll
      for (int r = 0; r < 8; ++r){ accP[r] = cg[(size_t)(r + mh) * HIDN + col] + tw; accT[r] = 0.0f; }
      v16bf b = ldB16(W1fL + ((j * 32 + lane) << 4));
      accP = wmma_bf16(aY, b, accP);
      accT = wmma_bf16(aE, b, accT);
#pragma unroll
      for (int r = 0; r < 8; ++r){
        float g, dg; gelu_both(accP[r], g, dg);
        const int off = (r + mh) * STRA + col;
        ab[off]  = f2bf(g);
        tab[off] = f2bf(accT[r] * dg);
      }
    }
    // ---- layer 2: hoist all 12 A-fragments to VGPRs (frees ab/tab for reuse)
    v16bf aP[6], aT[6];
#pragma unroll
    for (int kt = 0; kt < 6; ++kt){
      aP[kt] = ldA16(ab  + n * STRA + kt * 32 + mh);
      aT[kt] = ldA16(tab + n * STRA + kt * 32 + mh);
    }
#pragma unroll
    for (int j = 0; j < 12; ++j){
      const int col = (j << 4) + n;
      v8f accP, accT;
#pragma unroll
      for (int r = 0; r < 8; ++r){ accP[r] = b2L[col]; accT[r] = 0.0f; }
#pragma unroll
      for (int kt = 0; kt < 6; ++kt){
        v16bf b = ldB16(W2fL + (((j * 6 + kt) * 32 + lane) << 4));
        accP = wmma_bf16(aP[kt], b, accP);
        accT = wmma_bf16(aT[kt], b, accT);
      }
#pragma unroll
      for (int r = 0; r < 8; ++r){
        float g, dg; gelu_both(accP[r], g, dg);
        const int off = (r + mh) * STRA + col;
        ab[off]  = f2bf(g);               // layer-2 activation reuses layer-1 buffer
        tab[off] = f2bf(accT[r] * dg);
      }
    }
    // ---- layer 3: f = a2@W3 + b3 ; Je = ta2@W3   (C/D layout == RK4 layout)
    v8f accP, accT;
#pragma unroll
    for (int r = 0; r < 8; ++r){ accP[r] = b3L[n]; accT[r] = 0.0f; }
#pragma unroll
    for (int kt = 0; kt < 6; ++kt){
      v16bf a3 = ldA16(ab  + n * STRA + kt * 32 + mh);
      v16bf t3 = ldA16(tab + n * STRA + kt * 32 + mh);
      v16bf b  = ldB16(W3fL + ((kt * 32 + lane) << 4));
      accP = wmma_bf16(a3, b, accP);
      accT = wmma_bf16(t3, b, accT);
    }
#pragma unroll
    for (int r = 0; r < 8; ++r) fo[r] = accP[r];
    // ---- trace = sum_n e*Je per row (16-lane xor reduction, rows via LDS)
    float pr[8];
#pragma unroll
    for (int r = 0; r < 8; ++r) pr[r] = er[r] * accT[r];
#pragma unroll
    for (int msk = 1; msk <= 8; msk <<= 1)
#pragma unroll
      for (int r = 0; r < 8; ++r) pr[r] += __shfl_xor(pr[r], msk, 32);
    if (lane == 0){
#pragma unroll
      for (int r = 0; r < 8; ++r) red[r] = pr[r];
    }
    if (lane == 16){
#pragma unroll
      for (int r = 0; r < 8; ++r) red[8 + r] = pr[r];
    }
    ln = -red[n];
  };

#pragma unroll 1
  for (int step = 0; step < 10; ++step){
    const float t0 = 1.0f + dtv * (float)step;
    float lsum = 0.0f;
#pragma unroll
    for (int r = 0; r < 8; ++r) fsum[r] = 0.0f;
#pragma unroll 1
    for (int st = 0; st < 4; ++st){
      // torchdiffeq 3/8-rule stage coefficients (uniform selects, no arrays)
      const float ca1 = (st == 1) ? dtv * (1.0f / 3.0f) : (st == 2) ? -dtv * (1.0f / 3.0f)
                       : (st == 3) ? dtv : 0.0f;
      const float ca2 = (st == 2) ? dtv : (st == 3) ? -dtv : 0.0f;
      const float ca3 = (st == 3) ? dtv : 0.0f;
      const float tof = (st == 1) ? (1.0f / 3.0f) : (st == 2) ? (2.0f / 3.0f)
                       : (st == 3) ? 1.0f : 0.0f;
      const float wst = (st == 1 || st == 2) ? 3.0f : 1.0f;
#pragma unroll
      for (int r = 0; r < 8; ++r)
        ys[r] = y[r] + ca1 * f1[r] + ca2 * f2s[r] + ca3 * f3[r];
      {
        const int sidx = step * 4 + st;
        const float* ep = eps + ((size_t)sidx * B + rowbase) * DD;
#pragma unroll
        for (int r = 0; r < 8; ++r) er[r] = ep[(size_t)(r + mh) * DD + n];
        if (sidx + 1 < 40){  // prefetch next stage's noise slice (pure HBM stream)
          const float* ep2 = eps + ((size_t)(sidx + 1) * B + rowbase) * DD;
          __builtin_prefetch(ep2 + (size_t)mh * DD + n, 0, 1);
        }
      }
      float ln;
      vf(t0 + dtv * tof, ln);
#pragma unroll
      for (int r = 0; r < 8; ++r){
        fsum[r] = fmaf(wst, fo[r], fsum[r]);
        f1[r]  = (st == 0) ? fo[r] : f1[r];
        f2s[r] = (st == 1) ? fo[r] : f2s[r];
        f3[r]  = (st == 2) ? fo[r] : f3[r];
      }
      lsum = fmaf(wst, ln, lsum);
    }
#pragma unroll
    for (int r = 0; r < 8; ++r) y[r] = fmaf(dtv * 0.125f, fsum[r], y[r]);
    lp = fmaf(dtv * 0.125f, lsum, lp);
  }

  float pr[8];
#pragma unroll
  for (int r = 0; r < 8; ++r) pr[r] = y[r] * y[r];
#pragma unroll
  for (int msk = 1; msk <= 8; msk <<= 1)
#pragma unroll
    for (int r = 0; r < 8; ++r) pr[r] += __shfl_xor(pr[r], msk, 32);
  if (lane == 0){
#pragma unroll
    for (int r = 0; r < 8; ++r) red[r] = pr[r];
  }
  if (lane == 16){
#pragma unroll
    for (int r = 0; r < 8; ++r) red[8 + r] = pr[r];
  }
  float s = red[n];
  if (lane < 16) out[rowbase + lane] = -0.5f * s - 14.703016531274762f - lp;
}

extern "C" void kernel_launch(void* const* d_in, const int* in_sizes, int n_in,
                              void* d_out, int out_size, void* d_ws, size_t ws_size,
                              hipStream_t stream){
  const float* theta = (const float*)d_in[0];
  const float* h     = (const float*)d_in[1];
  const float* eps   = (const float*)d_in[2];
  const float* W1    = (const float*)d_in[3];
  const float* b1    = (const float*)d_in[4];
  const float* W2    = (const float*)d_in[5];
  const float* b2    = (const float*)d_in[6];
  const float* W3    = (const float*)d_in[7];
  const float* b3    = (const float*)d_in[8];
  float* out = (float*)d_out;
  const int B = in_sizes[0] / 16;

  float*          c    = (float*)d_ws;                                    // B*192 f32
  unsigned short* W2f  = (unsigned short*)((char*)d_ws + (size_t)B * 192 * 4);
  unsigned short* W1hf = W2f + 72 * 32 * 16;

  cnf_permute<<<dim3(96),      dim3(256), 0, stream>>>(W1, W2, W1hf, W2f);
  cnf_context<<<dim3(B / 128), dim3(256), 0, stream>>>(h, b1, W1hf, c);
  cnf_main<<<dim3(B / 64),     dim3(128), 0, stream>>>(theta, eps, W1, b2, W3, b3, c, W2f, out, B);
}
